// AttentionDecoder_29970281792131
// MI455X (gfx1250) — matrix operand order repaired and checked by reading, not compiled, not run on hardware
//
#include <hip/hip_runtime.h>
#include <math.h>

typedef __attribute__((ext_vector_type(16))) _Float16 v16h;
typedef __attribute__((ext_vector_type(8)))  float    v8f;

#define BB 32
#define NNODES 10000
#define DD 128
#define CH 20          // chunks of 512 rows over N
#define BLK_ROWS 512
#define NEGPEN (-1e9f)

// ---------------------------------------------------------------------------
// Load 4 WMMA A-fragments (16x32 f16) covering rows [base,base+16) x K=0..127 of
// node_embed (fp32 row-major [N][128]), f32->f16 on the fly.
// 16-bit A 16x32 in this code's contiguous k order (not the ISA table order): lane L holds row M=L&15;
// lanes 0-15: K=0..15 contiguous; lanes 16-31: K=16..31.
// ---------------------------------------------------------------------------
__device__ __forceinline__ void load_a_frags(v16h a[4], const float* __restrict__ eb,
                                             int base, int lane) {
  int row = base + (lane & 15);
  if (row >= NNODES) row = NNODES - 1;   // clamped; zero-weighted later
  const float* rp = eb + (size_t)row * DD;
  const int kext = (lane & 16) ? 16 : 0;
#pragma unroll
  for (int kk = 0; kk < 4; kk++) {
#pragma unroll
    for (int p = 0; p < 8; p++) {
      int kOff = kk * 32 + kext + 2 * p;
      float2 v = *(const float2*)(rp + kOff);
      a[kk][2 * p]     = (_Float16)v.x;
      a[kk][2 * p + 1] = (_Float16)v.y;
    }
  }
}

// ---------------------------------------------------------------------------
// K1: deterministic partial mean: gpart[ch][b][c] = sum of 100 rows, column c
// ---------------------------------------------------------------------------
__global__ __launch_bounds__(128) void k_mean(const float* __restrict__ ne,
                                              float* __restrict__ gpart) {
  int ch = blockIdx.x, b = blockIdx.y, c = threadIdx.x;
  const float* eb = ne + (size_t)b * NNODES * DD;
  float s = 0.f;
  int r0 = ch * 100;
  for (int r = 0; r < 100; r++) s += eb[(size_t)(r0 + r) * DD + c];
  gpart[((size_t)ch * BB + b) * DD + c] = s;
}

// ---------------------------------------------------------------------------
// K2: query q[b] = mean@W_fixed + concat(e_first,e_last)@W_step, then fold q into
// the K-projection:  P[b][h][k] = sum_d Wk[k][h*16+d] * q[h*16+d]   (128x8)
// so compat = (E @ P) exactly.
// ---------------------------------------------------------------------------
__global__ __launch_bounds__(128) void k_query(const float* __restrict__ ne,
                                               const float* __restrict__ gpart,
                                               const float* __restrict__ W_fixed,
                                               const float* __restrict__ W_proj,
                                               const float* __restrict__ W_step,
                                               const int* __restrict__ first_node,
                                               const int* __restrict__ last_node,
                                               float* __restrict__ Pg) {
  int b = blockIdx.x, j = threadIdx.x;
  __shared__ float ge[128], sc[256], qs[128];
  float s = 0.f;
  for (int ch = 0; ch < 100; ch++) s += gpart[((size_t)ch * BB + b) * DD + j];
  ge[j] = s * (1.0f / (float)NNODES);
  int fn = first_node[b], ln_ = last_node[b];
  const float* eb = ne + (size_t)b * NNODES * DD;
  sc[j]       = eb[(size_t)fn * DD + j];
  sc[128 + j] = eb[(size_t)ln_ * DD + j];
  __syncthreads();
  float acc = 0.f;
  for (int k = 0; k < 128; k++) acc += ge[k] * W_fixed[k * 128 + j];
  for (int k = 0; k < 256; k++) acc += sc[k] * W_step[k * 128 + j];
  qs[j] = acc;
  __syncthreads();
  for (int idx = j; idx < 1024; idx += 128) {
    int h = idx >> 7, k = idx & 127;
    float p = 0.f;
#pragma unroll
    for (int d = 0; d < 16; d++) p += W_proj[(size_t)k * 384 + h * 16 + d] * qs[h * 16 + d];
    Pg[(size_t)b * 1024 + h * 128 + k] = p;
  }
}

// ---------------------------------------------------------------------------
// K3: flash glimpse-attention. compat = E@P (4 WMMAs / 16 rows, head==lane).
// z_h = sum_row w_h[row]*e_row accumulated via WMMA rank-32 updates
// (A = w matrix packed per-lane, B = E rows in B-layout, WGP$-hot reread).
// Per-block LSE partial (m,l,z[128]) per head -> attnP.  No weights in hot loop.
// ---------------------------------------------------------------------------
__global__ __launch_bounds__(256) void k_attn(const float* __restrict__ ne,
                                              const float* __restrict__ Pg,
                                              const float* __restrict__ mask,
                                              const float* __restrict__ gmask,
                                              float* __restrict__ attnP) {
  __shared__ float red[8][8][130];   // [wave][head][m,l,z0..z127] = 33.3 KB
  const int chunk = blockIdx.x, b = blockIdx.y;
  const int tid = threadIdx.x, lane = tid & 31, wv = tid >> 5;
  const int head = lane & 15;
  const int upper = (lane & 16) ? 1 : 0;
  const int mext = upper ? 8 : 0;

  // P as 4 B-fragments (32x16, cols 8..15 zero-padded), kept in registers
  v16h pf[4];
  {
    const float* Pb = Pg + (size_t)b * 1024;
#pragma unroll
    for (int kk = 0; kk < 4; kk++)
#pragma unroll
      for (int e = 0; e < 16; e++)
        pf[kk][e] = (head < 8)
                  ? (_Float16)Pb[head * 128 + kk * 32 + e + (upper ? 16 : 0)]
                  : (_Float16)0.f;
  }
  const float* eb   = ne + (size_t)b * NNODES * DD;
  const float* mrow = mask + (size_t)b * NNODES;
  const float* grow = gmask + (size_t)b * NNODES;

  float m = -1e30f, l = 0.f;
  v8f S[8];
#pragma unroll
  for (int nn = 0; nn < 8; nn++) S[nn] = (v8f){0.f,0.f,0.f,0.f,0.f,0.f,0.f,0.f};

  for (int pr = 0; pr < 2; pr++) {               // 2 pairs of 32 rows per wave
    int pairBase = chunk * BLK_ROWS + wv * 64 + pr * 32;
    if (pairBase >= NNODES) continue;            // uniform per wave, no syncs inside
    float cA[8], cB[8];
    bool  vA[8], vB[8];
#pragma unroll
    for (int t2 = 0; t2 < 2; t2++) {
      int base = pairBase + t2 * 16;
      v16h a[4];
      load_a_frags(a, eb, base, lane);
      v8f c = {0.f,0.f,0.f,0.f,0.f,0.f,0.f,0.f};
#pragma unroll
      for (int kk = 0; kk < 4; kk++)
        c = __builtin_amdgcn_wmma_f32_16x16x32_f16(false, a[kk], false, pf[kk],
                                                   (short)0, c, false, false);
      float penL = 0.f;
      if (lane < 16) {
        int rw = base + lane;
        if (rw < NNODES) penL = (mrow[rw] + grow[rw]) * NEGPEN;
      }
#pragma unroll
      for (int r = 0; r < 8; r++) {
        bool v = (base + r + mext) < NNODES;
        float cv = v ? (c[r] * 0.25f + __shfl(penL, r + mext)) : -1e30f;
        if (t2 == 0) { cA[r] = cv; vA[r] = v; } else { cB[r] = cv; vB[r] = v; }
      }
    }
    // per-lane (== per-head) online softmax over the 32 rows
    float tm = -1e30f;
#pragma unroll
    for (int r = 0; r < 8; r++) tm = fmaxf(tm, fmaxf(cA[r], cB[r]));
    tm = fmaxf(tm, __shfl_xor(tm, 16));
    float mnew  = fmaxf(m, tm);
    float scale = __expf(m - mnew);
    m = mnew;
    float wA[8], wB[8], wsum = 0.f;
    bool hv = head < 8;
#pragma unroll
    for (int r = 0; r < 8; r++) {
      wA[r] = (hv && vA[r]) ? __expf(cA[r] - mnew) : 0.f;
      wB[r] = (hv && vB[r]) ? __expf(cB[r] - mnew) : 0.f;
      wsum += wA[r] + wB[r];
    }
    l = l * scale + wsum + __shfl_xor(wsum, 16);
    // rescale z rows (row==head in S fragments) by per-head scale
#pragma unroll
    for (int r = 0; r < 8; r++) {
      float scR = __shfl(scale, r + mext);
#pragma unroll
      for (int nn = 0; nn < 8; nn++) S[nn][r] *= scR;
    }
    // pack w (8x32) into an A-fragment: pure per-lane packing
    v16h aw;
#pragma unroll
    for (int e = 0; e < 8; e++) {
      aw[e]     = (_Float16)wA[e];
      aw[8 + e] = (_Float16)wB[e];
    }
    // z += w @ E_pair : 8 WMMAs, E re-gathered in B layout (L0/WGP$ hot)
#pragma unroll
    for (int nn = 0; nn < 8; nn++) {
      v16h ebf;
      int col = nn * 16 + head;
#pragma unroll
      for (int e = 0; e < 16; e++) {
        int row = pairBase + e + (upper ? 8 : 0) + ((e >= 8) ? 8 : 0);
        if (row >= NNODES) row = NNODES - 1;     // weight is 0 there
        ebf[e] = (_Float16)eb[(size_t)row * DD + col];
      }
      S[nn] = __builtin_amdgcn_wmma_f32_16x16x32_f16(false, aw, false, ebf,
                                                     (short)0, S[nn], false, false);
    }
  }
  // wave partial -> LDS (heads 0..7 live in VGPR rows 0..7, lower lanes)
  if (lane < 8) { red[wv][lane][0] = m; red[wv][lane][1] = l; }
  if (lane < 16) {
#pragma unroll
    for (int nn = 0; nn < 8; nn++)
#pragma unroll
      for (int r = 0; r < 8; r++)
        red[wv][r][2 + nn * 16 + lane] = S[nn][r];
  }
  __syncthreads();
  // block-level LSE merge of 8 wave partials
  for (int idx = tid; idx < 1024; idx += 256) {
    int h = idx >> 7, c = idx & 127;
    float M = -1e30f;
#pragma unroll
    for (int w8 = 0; w8 < 8; w8++) M = fmaxf(M, red[w8][h][0]);
    float Z = 0.f, L = 0.f;
#pragma unroll
    for (int w8 = 0; w8 < 8; w8++) {
      float e = __expf(red[w8][h][0] - M);
      Z += e * red[w8][h][2 + c];
      L += e * red[w8][h][1];
    }
    float* outp = attnP + (((size_t)b * CH + chunk) * 8 + h) * 130;
    if (c == 0) { outp[0] = M; outp[1] = L; }
    outp[2 + c] = Z;
  }
}

// ---------------------------------------------------------------------------
// K4: merge chunk partials -> z,l per head; heads = (z/l)@Wv_h; glimpse = heads@W_out;
// fold glimpse into logit-K:  u[b][k] = sum_j Wl[k][j]*glimpse[j]
// ---------------------------------------------------------------------------
__global__ __launch_bounds__(128) void k_merge(const float* __restrict__ attnP,
                                               const float* __restrict__ W_proj,
                                               const float* __restrict__ W_out,
                                               float* __restrict__ ug) {
  const int b = blockIdx.x, tid = threadIdx.x;
  __shared__ float zf[8][128], lh[8], hv[128], gl[128];
  for (int idx = tid; idx < 1024; idx += 128) {
    int h = idx >> 7, c = idx & 127;
    float M = -1e30f;
    for (int ch = 0; ch < CH; ch++)
      M = fmaxf(M, attnP[(((size_t)b * CH + ch) * 8 + h) * 130]);
    float Z = 0.f, L = 0.f;
    for (int ch = 0; ch < CH; ch++) {
      const float* p = attnP + (((size_t)b * CH + ch) * 8 + h) * 130;
      float e = __expf(p[0] - M);
      Z += e * p[2 + c];
      L += e * p[1];
    }
    zf[h][c] = Z;
    if (c == 0) lh[h] = L;
  }
  __syncthreads();
  {
    int h = tid >> 4, d = tid & 15;
    float s = 0.f;
    for (int k = 0; k < 128; k++)
      s += zf[h][k] * W_proj[(size_t)k * 384 + 128 + h * 16 + d];
    hv[tid] = s / lh[h];
  }
  __syncthreads();
  {
    float g = 0.f;
    for (int k = 0; k < 128; k++) g += hv[k] * W_out[k * 128 + tid];
    gl[tid] = g;
  }
  __syncthreads();
  {
    float uu = 0.f;
    for (int jj = 0; jj < 128; jj++) uu += W_proj[(size_t)tid * 384 + 256 + jj] * gl[jj];
    ug[b * 128 + tid] = uu;
  }
}

// ---------------------------------------------------------------------------
// K5: logits = (E @ u)/sqrt(128), tanh-clipped + masked; per-chunk LSE partials.
// ---------------------------------------------------------------------------
__global__ __launch_bounds__(256) void k_logits(const float* __restrict__ ne,
                                                const float* __restrict__ ug,
                                                const float* __restrict__ mask,
                                                const float* __restrict__ gmask,
                                                float* __restrict__ tbuf,
                                                float* __restrict__ lseP) {
  __shared__ float smM[8], smL[8];
  const int chunk = blockIdx.x, b = blockIdx.y;
  const int tid = threadIdx.x, lane = tid & 31, wv = tid >> 5;
  const int upper = (lane & 16) ? 1 : 0;
  const int mext = upper ? 8 : 0;
  v16h uf[4];
#pragma unroll
  for (int kk = 0; kk < 4; kk++)
#pragma unroll
    for (int e = 0; e < 16; e++)
      uf[kk][e] = ((lane & 15) == 0)
                ? (_Float16)ug[b * 128 + kk * 32 + e + (upper ? 16 : 0)]
                : (_Float16)0.f;
  const float* eb   = ne + (size_t)b * NNODES * DD;
  const float* mrow = mask + (size_t)b * NNODES;
  const float* grow = gmask + (size_t)b * NNODES;
  float lm = -1e30f, ll = 0.f;

  for (int t = 0; t < 4; t++) {
    int base = chunk * BLK_ROWS + wv * 64 + t * 16;
    if (base >= NNODES) break;
    v16h a[4];
    load_a_frags(a, eb, base, lane);
    v8f c = {0.f,0.f,0.f,0.f,0.f,0.f,0.f,0.f};
#pragma unroll
    for (int kk = 0; kk < 4; kk++)
      c = __builtin_amdgcn_wmma_f32_16x16x32_f16(false, a[kk], false, uf[kk],
                                                 (short)0, c, false, false);
    if ((lane & 15) == 0) {                       // column 0 holds the dot products
#pragma unroll
      for (int r = 0; r < 8; r++) {
        int row = base + r + mext;
        if (row < NNODES) {
          float lg = c[r] * 0.08838834764831845f;  // 1/sqrt(128)
          lg += grow[row] * NEGPEN;
          lg = tanhf(lg) * 10.f;
          lg += mrow[row] * NEGPEN;
          tbuf[(size_t)b * NNODES + row] = lg;
          float nm = fmaxf(lm, lg);
          ll = ll * __expf(lm - nm) + __expf(lg - nm);
          lm = nm;
        }
      }
    }
  }
#pragma unroll
  for (int msk = 1; msk <= 16; msk <<= 1) {
    float om = __shfl_xor(lm, msk), ol = __shfl_xor(ll, msk);
    float M2 = fmaxf(lm, om);
    ll = ll * __expf(lm - M2) + ol * __expf(om - M2);
    lm = M2;
  }
  if (lane == 0) { smM[wv] = lm; smL[wv] = ll; }
  __syncthreads();
  if (tid == 0) {
    float M2 = -1e30f, L2 = 0.f;
    for (int w = 0; w < 8; w++) M2 = fmaxf(M2, smM[w]);
    for (int w = 0; w < 8; w++) L2 += smL[w] * __expf(smM[w] - M2);
    lseP[((size_t)b * CH + chunk) * 2]     = M2;
    lseP[((size_t)b * CH + chunk) * 2 + 1] = L2;
  }
}

__global__ void k_lse(const float* __restrict__ lseP, float* __restrict__ lse) {
  int b = blockIdx.x;
  if (threadIdx.x == 0) {
    float M = -1e30f, L = 0.f;
    for (int c = 0; c < CH; c++) M = fmaxf(M, lseP[((size_t)b * CH + c) * 2]);
    for (int c = 0; c < CH; c++)
      L += lseP[((size_t)b * CH + c) * 2 + 1] * __expf(lseP[((size_t)b * CH + c) * 2] - M);
    lse[b] = M + logf(L);
  }
}

__global__ void k_out(const float* __restrict__ tbuf, const float* __restrict__ lse,
                      float* __restrict__ out) {
  int i = blockIdx.x * blockDim.x + threadIdx.x;
  if (i < BB * NNODES) out[i] = tbuf[i] - lse[i / NNODES];
}

// ---------------------------------------------------------------------------
extern "C" void kernel_launch(void* const* d_in, const int* in_sizes, int n_in,
                              void* d_out, int out_size, void* d_ws, size_t ws_size,
                              hipStream_t stream) {
  const float* node_embed = (const float*)d_in[0];
  const float* W_fixed    = (const float*)d_in[1];
  const float* W_proj     = (const float*)d_in[2];
  const float* W_step     = (const float*)d_in[3];
  const float* W_out      = (const float*)d_in[4];
  const int*   first_node = (const int*)d_in[5];
  const int*   last_node  = (const int*)d_in[6];
  const float* mask       = (const float*)d_in[7];
  const float* gmask      = (const float*)d_in[8];
  float* out = (float*)d_out;

  float* wsf   = (float*)d_ws;
  float* gpart = wsf;                                    // 100*32*128 = 409600
  float* Pg    = gpart + 100 * BB * DD;                  // 32*1024   = 32768
  float* attnP = Pg + (size_t)BB * 1024;                 // 32*20*8*130 = 665600
  float* ug    = attnP + (size_t)BB * CH * 8 * 130;      // 4096
  float* tbuf  = ug + BB * DD;                           // 320000
  float* lseP  = tbuf + (size_t)BB * NNODES;             // 1280
  float* lse   = lseP + BB * CH * 2;                     // 32

  k_mean  <<<dim3(100, BB), 128, 0, stream>>>(node_embed, gpart);
  k_query <<<BB, 128, 0, stream>>>(node_embed, gpart, W_fixed, W_proj, W_step,
                                   first_node, last_node, Pg);
  k_attn  <<<dim3(CH, BB), 256, 0, stream>>>(node_embed, Pg, mask, gmask, attnP);
  k_merge <<<BB, 128, 0, stream>>>(attnP, W_proj, W_out, ug);
  k_logits<<<dim3(CH, BB), 256, 0, stream>>>(node_embed, ug, mask, gmask, tbuf, lseP);
  k_lse   <<<BB, 32, 0, stream>>>(lseP, lse);
  k_out   <<<(BB * NNODES + 255) / 256, 256, 0, stream>>>(tbuf, lse, out);
}
